// CompPCFG_32083405701426
// MI455X (gfx1250) — matrix operand order repaired and checked by reading, not compiled, not run on hardware
//
#include <hip/hip_runtime.h>

// ---------------- problem constants ----------------
#define BB   16      // batch
#define NSEQ 30      // sequence length n
#define NTC  30      // nonterminals
#define TTC  60      // terminals
#define DD   256     // hidden
#define ZZ   64      // latent
#define VV   10000   // vocab
#define VVP  10016   // vocab padded to mult of 32
#define SSY  90      // S = NT + T
#define SPAD 96      // S padded to mult of 32 (WMMA K)
#define S2   8100    // S*S
#define S2P  8128    // S*S padded to mult of 32
#define DZK  320     // D + Z (mult of 32)
#define NTS  2700    // NT*S
#define NTSP 2720    // padded (= 32*85)
#define RMAX 256     // max padded rows per b in inside stage (w=16: 225 -> 256)
#define NEGV -1000000000.0f

// ---------------- WMMA types / helpers ----------------
typedef __attribute__((ext_vector_type(16))) __bf16 bf16x16;
typedef __attribute__((ext_vector_type(8)))  float  floatx8;

union BFrag { bf16x16 v; unsigned short u[16]; };

__device__ __forceinline__ unsigned short f2bf(float f) {
    union { float f; unsigned u; } a; a.f = f;
    unsigned r = a.u + 0x7FFFu + ((a.u >> 16) & 1u);   // round-to-nearest-even
    return (unsigned short)(r >> 16);
}
__device__ __forceinline__ float bf2f(unsigned short h) {
    union { unsigned u; float f; } a; a.u = ((unsigned)h) << 16;
    return a.f;
}

// A 16x32 bf16 fragment: lanes 0-15 rows, K {0..7,16..23}; lanes 16-31 K {8..15,24..31}
__device__ __forceinline__ void load_afrag(BFrag& a, const unsigned short* Ap, int half) {
#pragma unroll
    for (int i = 0; i < 8; ++i) {
        const int ka = (i < 4) ? (half * 8 + 2 * i) : (16 + half * 8 + 2 * (i - 4));
        a.u[2 * i]     = Ap[ka];
        a.u[2 * i + 1] = Ap[ka + 1];
    }
}
// B 32x16 bf16 fragment (W row-major, N x K): lanes 0-15 cols, K {0..7,16..23}; lanes 16-31 K {8..15,24..31}
__device__ __forceinline__ void load_bfrag(BFrag& b, const unsigned short* Wp, int half) {
#pragma unroll
    for (int i = 0; i < 8; ++i) {
        const int kb = (i < 4) ? (half * 8 + 2 * i) : (16 + half * 8 + 2 * (i - 4));
        b.u[2 * i]     = Wp[kb];
        b.u[2 * i + 1] = Wp[kb + 1];
    }
}

// ---------------------------------------------------------------------------
// bf16 WMMA GEMM, 16x16 tile per wave (fallback for ragged/small shapes):
//   C[M,N](f32) = act( A[M,K](bf16) @ W[N,K](bf16)^T + bias[N] ) (+res[M,N])
// ---------------------------------------------------------------------------
__global__ void k_gemm16(const unsigned short* __restrict__ A,
                         const unsigned short* __restrict__ W,
                         const float* __restrict__ bias,
                         const float* __restrict__ res,
                         float* __restrict__ C,
                         unsigned short* __restrict__ Cb,
                         int M, int N, int K, int relu,
                         long sA, long sW, long sC)
{
    const int bidx = blockIdx.y;
    A += (size_t)bidx * sA;  W += (size_t)bidx * sW;  C += (size_t)bidx * sC;

    const int wave = threadIdx.x >> 5;
    const int lane = threadIdx.x & 31;
    const int tilesN = N >> 4;
    const int tile = blockIdx.x * (blockDim.x >> 5) + wave;
    if (tile >= (M >> 4) * tilesN) return;
    const int m0 = (tile / tilesN) << 4;
    const int n0 = (tile % tilesN) << 4;
    const int half = lane >> 4;
    const int r16  = lane & 15;

    floatx8 acc = {0.f, 0.f, 0.f, 0.f, 0.f, 0.f, 0.f, 0.f};
    const unsigned short* Ap = A + (size_t)(m0 + r16) * K;
    const unsigned short* Wp = W + (size_t)(n0 + r16) * K;

    for (int kk = 0; kk < K; kk += 32) {
        BFrag a, b;
        load_afrag(a, Ap + kk, half);
        load_bfrag(b, Wp + kk, half);
        acc = __builtin_amdgcn_wmma_f32_16x16x32_bf16(false, a.v, false, b.v,
                                                      (short)0, acc, false, false);
    }

    const int nc = n0 + r16;
#pragma unroll
    for (int j = 0; j < 8; ++j) {
        const int mc = m0 + j + 8 * half;
        float v = acc[j];
        if (bias) v += bias[nc];
        if (relu) v = fmaxf(v, 0.f);
        if (res)  v += res[(size_t)mc * N + nc];    // relu BEFORE residual add
        C[(size_t)mc * N + nc] = v;
        if (Cb) Cb[(size_t)mc * N + nc] = f2bf(v);
    }
}

// ---------------------------------------------------------------------------
// bf16 WMMA GEMM, 32x32 macro-tile per wave (M%32==0, N%32==0):
// 2 A-frags x 2 B-frags -> 4 WMMAs per 32-K chunk (2x arithmetic intensity).
// ---------------------------------------------------------------------------
__global__ void k_gemm32(const unsigned short* __restrict__ A,
                         const unsigned short* __restrict__ W,
                         const float* __restrict__ bias,
                         const float* __restrict__ res,
                         float* __restrict__ C,
                         unsigned short* __restrict__ Cb,
                         int M, int N, int K, int relu,
                         long sA, long sW, long sC)
{
    const int bidx = blockIdx.y;
    A += (size_t)bidx * sA;  W += (size_t)bidx * sW;  C += (size_t)bidx * sC;

    const int wave = threadIdx.x >> 5;
    const int lane = threadIdx.x & 31;
    const int tilesN = N >> 5;
    const int tile = blockIdx.x * (blockDim.x >> 5) + wave;
    if (tile >= (M >> 5) * tilesN) return;
    const int m0 = (tile / tilesN) << 5;
    const int n0 = (tile % tilesN) << 5;
    const int half = lane >> 4;
    const int r16  = lane & 15;

    floatx8 acc00 = {0.f,0.f,0.f,0.f,0.f,0.f,0.f,0.f};
    floatx8 acc01 = acc00, acc10 = acc00, acc11 = acc00;

    const unsigned short* Ap0 = A + (size_t)(m0 + r16) * K;
    const unsigned short* Ap1 = A + (size_t)(m0 + 16 + r16) * K;
    const unsigned short* Wp0 = W + (size_t)(n0 + r16) * K;
    const unsigned short* Wp1 = W + (size_t)(n0 + 16 + r16) * K;

    for (int kk = 0; kk < K; kk += 32) {
        BFrag a0, a1, b0, b1;
        load_afrag(a0, Ap0 + kk, half);
        load_afrag(a1, Ap1 + kk, half);
        load_bfrag(b0, Wp0 + kk, half);
        load_bfrag(b1, Wp1 + kk, half);
        __builtin_prefetch(Wp0 + kk + 64, 0, 0);
        acc00 = __builtin_amdgcn_wmma_f32_16x16x32_bf16(false, a0.v, false, b0.v, (short)0, acc00, false, false);
        acc01 = __builtin_amdgcn_wmma_f32_16x16x32_bf16(false, a0.v, false, b1.v, (short)0, acc01, false, false);
        acc10 = __builtin_amdgcn_wmma_f32_16x16x32_bf16(false, a1.v, false, b0.v, (short)0, acc10, false, false);
        acc11 = __builtin_amdgcn_wmma_f32_16x16x32_bf16(false, a1.v, false, b1.v, (short)0, acc11, false, false);
    }

#pragma unroll
    for (int ti = 0; ti < 2; ++ti) {
#pragma unroll
        for (int tj = 0; tj < 2; ++tj) {
            const floatx8 acc = (ti == 0) ? (tj == 0 ? acc00 : acc01)
                                          : (tj == 0 ? acc10 : acc11);
            const int nc = n0 + tj * 16 + r16;
#pragma unroll
            for (int j = 0; j < 8; ++j) {
                const int mc = m0 + ti * 16 + j + 8 * half;
                float v = acc[j];
                if (bias) v += bias[nc];
                if (relu) v = fmaxf(v, 0.f);
                if (res)  v += res[(size_t)mc * N + nc];
                C[(size_t)mc * N + nc] = v;
                if (Cb) Cb[(size_t)mc * N + nc] = f2bf(v);
            }
        }
    }
}

// ---------------- packing / setup kernels ----------------
__global__ void k_pack(const float* __restrict__ src, unsigned short* __restrict__ dst,
                       int Rs, int Cs, int Rp, int Cp)
{
    long i = (long)blockIdx.x * blockDim.x + threadIdx.x;
    if (i >= (long)Rp * Cp) return;
    int r = (int)(i / Cp), c = (int)(i % Cp);
    float v = (r < Rs && c < Cs) ? src[(size_t)r * Cs + c] : 0.f;
    dst[i] = f2bf(v);
}

__global__ void k_padf(const float* __restrict__ src, float* __restrict__ dst, int Cs, int Cp)
{
    int i = blockIdx.x * blockDim.x + threadIdx.x;
    if (i >= Cp) return;
    dst[i] = (i < Cs) ? src[i] : 0.f;
}

// X = [emb_row | z_b]  (rows = B*R, cols = 320), bf16
__global__ void k_build_x(const float* __restrict__ emb, const float* __restrict__ z,
                          unsigned short* __restrict__ dst, int R)
{
    long i = (long)blockIdx.x * blockDim.x + threadIdx.x;
    long rows = (long)BB * R;
    if (i >= rows * DZK) return;
    int row = (int)(i / DZK), c = (int)(i % DZK);
    int b = row / R, r = row % R;
    float v = (c < DD) ? emb[(size_t)r * DD + c] : z[(size_t)b * ZZ + (c - DD)];
    dst[i] = f2bf(v);
}

// per-row max and logsumexp
__global__ void k_rowstat(const float* __restrict__ X, float* __restrict__ rmax,
                          float* __restrict__ rlse, int Nv, int stride)
{
    const int row = blockIdx.x;
    const float* p = X + (size_t)row * stride;
    __shared__ float sm[256];
    float m = -3.4e38f;
    for (int c = threadIdx.x; c < Nv; c += blockDim.x) m = fmaxf(m, p[c]);
    sm[threadIdx.x] = m; __syncthreads();
    for (int s = blockDim.x >> 1; s > 0; s >>= 1) {
        if ((int)threadIdx.x < s) sm[threadIdx.x] = fmaxf(sm[threadIdx.x], sm[threadIdx.x + s]);
        __syncthreads();
    }
    m = sm[0]; __syncthreads();
    float acc = 0.f;
    for (int c = threadIdx.x; c < Nv; c += blockDim.x) acc += __expf(p[c] - m);
    sm[threadIdx.x] = acc; __syncthreads();
    for (int s = blockDim.x >> 1; s > 0; s >>= 1) {
        if ((int)threadIdx.x < s) sm[threadIdx.x] += sm[threadIdx.x + s];
        __syncthreads();
    }
    if (threadIdx.x == 0) { rmax[row] = m; rlse[row] = m + __logf(sm[0]); }
}

// unary[b,i,t] = vocab_logit[b*T+t, trg[b,i]] - lse[b*T+t]
__global__ void k_unary(const float* __restrict__ logv, const float* __restrict__ lsev,
                        const int* __restrict__ trg, float* __restrict__ unary)
{
    int i = blockIdx.x * blockDim.x + threadIdx.x;
    if (i >= BB * NSEQ * TTC) return;
    int t = i % TTC, pos = (i / TTC) % NSEQ, b = i / (TTC * NSEQ);
    int row = b * TTC + t;
    int v = trg[b * NSEQ + pos];
    unary[i] = logv[(size_t)row * VVP + v] - lsev[row];
}

__global__ void k_rootls(const float* __restrict__ logr, const float* __restrict__ lser,
                         float* __restrict__ rls)
{
    int i = blockIdx.x * blockDim.x + threadIdx.x;
    if (i >= BB * 32) return;
    int b = i / 32, k = i % 32;
    rls[i] = (k < NTC) ? (logr[(size_t)b * 32 + k] - lser[b]) : NEGV;
}

__global__ void k_km(const float* __restrict__ rmax, const float* __restrict__ rlse,
                     float* __restrict__ km)
{
    int i = blockIdx.x * blockDim.x + threadIdx.x;
    if (i < BB * NTC) km[i] = rmax[i] - rlse[i];
}

__global__ void k_zero16(unsigned short* p, long n)
{
    long i = (long)blockIdx.x * blockDim.x + threadIdx.x;
    if (i < n) p[i] = 0;
}

// erule packed for stage-1 GEMM: [b][(k*S+l)][r] bf16, = exp(logit - rowmax)
__global__ void k_erule(const float* __restrict__ logrule, const float* __restrict__ rmax,
                        unsigned short* __restrict__ erule)
{
    long i = (long)blockIdx.x * blockDim.x + threadIdx.x;
    if (i >= (long)BB * NTC * S2) return;
    int c = (int)(i % S2);
    int row = (int)(i / S2);                 // b*NT + k
    int b = row / NTC, k = row % NTC;
    int l = c / SSY, r = c % SSY;
    float e = __expf(logrule[(size_t)row * S2P + c] - rmax[row]);
    erule[((size_t)b * NTSP + (k * SSY + l)) * SPAD + r] = f2bf(e);
}

__global__ void k_beta_fill(float* __restrict__ beta)
{
    long i = (long)blockIdx.x * blockDim.x + threadIdx.x;
    if (i < (long)BB * NSEQ * NSEQ * SPAD) beta[i] = NEGV;
}

__global__ void k_beta_diag(const float* __restrict__ unary, float* __restrict__ beta)
{
    int i = blockIdx.x * blockDim.x + threadIdx.x;
    if (i >= BB * NSEQ * TTC) return;
    int t = i % TTC, pos = (i / TTC) % NSEQ, b = i / (TTC * NSEQ);
    beta[(((size_t)b * NSEQ + pos) * NSEQ + pos) * SPAD + NTC + t] = unary[i];
}

// per (b, span, split): EL/ER = exp(child - max), bf16, padded rows zeroed
__global__ void k_prep(const float* __restrict__ beta,
                       unsigned short* __restrict__ EL, unsigned short* __restrict__ ER,
                       float* __restrict__ lm, float* __restrict__ rm,
                       int w, int rowsPad)
{
    int i = blockIdx.x * blockDim.x + threadIdx.x;
    if (i >= BB * rowsPad) return;
    int b = i / rowsPad, row = i % rowsPad;
    int ns = NSEQ - w + 1;
    int rowsReal = ns * (w - 1);
    unsigned short* el = EL + (size_t)i * SPAD;
    unsigned short* er = ER + (size_t)i * SPAD;
    if (row >= rowsReal) {
        for (int c = 0; c < SPAD; ++c) { el[c] = 0; er[c] = 0; }
        lm[i] = 0.f; rm[i] = 0.f;
        return;
    }
    int s = row / (w - 1), u = row % (w - 1);
    int le = s + u, re = s + w - 1;
    const float* L = beta + (((size_t)b * NSEQ + s)        * NSEQ + le) * SPAD;
    const float* R = beta + (((size_t)b * NSEQ + (le + 1)) * NSEQ + re) * SPAD;
    float ml = -3.4e38f, mr = -3.4e38f;
    for (int c = 0; c < SSY; ++c) { ml = fmaxf(ml, L[c]); mr = fmaxf(mr, R[c]); }
    for (int c = 0; c < SPAD; ++c) {
        float lv = (c < SSY) ? __expf(L[c] - ml) : 0.f;
        float rv = (c < SSY) ? __expf(R[c] - mr) : 0.f;
        el[c] = f2bf(lv); er[c] = f2bf(rv);
    }
    lm[i] = ml; rm[i] = mr;
}

// stage-2: t[k]=sum_l el[l]*Out1[row,k*S+l]; sc=log t+lm+rm+km; lse over splits u
__global__ void k_combine(const unsigned short* __restrict__ EL,
                          const float* __restrict__ lm, const float* __restrict__ rm,
                          const float* __restrict__ km, const float* __restrict__ Out1,
                          float* __restrict__ beta, int w, int rowsPad)
{
    int ns = NSEQ - w + 1;
    int b = blockIdx.x / ns, s = blockIdx.x % ns;
    int tid = threadIdx.x;
    int k = tid >> 3;                 // 8 lanes per k
    int l8 = tid & 7;
    if (k >= NTC) return;
    float m = -3.4e38f, ssum = 0.f;
    for (int u = 0; u < w - 1; ++u) {
        int row = s * (w - 1) + u;
        int gi = b * rowsPad + row;
        const unsigned short* el = EL + (size_t)gi * SPAD;
        const float* o = Out1 + ((size_t)b * rowsPad + row) * NTSP + k * SSY;
        float part = 0.f;
        for (int l = l8; l < SSY; l += 8) part += bf2f(el[l]) * o[l];
        part += __shfl_xor(part, 1);
        part += __shfl_xor(part, 2);
        part += __shfl_xor(part, 4);
        float sc = (part > 0.f) ? (__logf(part) + lm[gi] + rm[gi] + km[b * NTC + k]) : NEGV;
        if (sc > m) { ssum = ssum * __expf(m - sc) + 1.f; m = sc; }
        else        { ssum += __expf(sc - m); }
    }
    if (l8 == 0)
        beta[(((size_t)b * NSEQ + s) * NSEQ + (s + w - 1)) * SPAD + k] = m + __logf(ssum);
}

__global__ void k_final(const float* __restrict__ beta, const float* __restrict__ rls,
                        float* __restrict__ out)
{
    int b = threadIdx.x;
    if (b >= BB) return;
    const float* p = beta + (((size_t)b * NSEQ + 0) * NSEQ + (NSEQ - 1)) * SPAD;
    float v[NTC], m = -3.4e38f;
    for (int k = 0; k < NTC; ++k) { v[k] = p[k] + rls[b * 32 + k]; m = fmaxf(m, v[k]); }
    float s = 0.f;
    for (int k = 0; k < NTC; ++k) s += __expf(v[k] - m);
    out[b] = -(m + __logf(s));
    out[BB + b] = 0.f;   // kl
}

// ---------------------------------------------------------------------------
extern "C" void kernel_launch(void* const* d_in, const int* in_sizes, int n_in,
                              void* d_out, int out_size, void* d_ws, size_t ws_size,
                              hipStream_t stream)
{
    (void)in_sizes; (void)n_in; (void)out_size; (void)ws_size;
    const int*   trg      = (const int*)  d_in[0];
    const float* z        = (const float*)d_in[1];
    const float* t_emb    = (const float*)d_in[2];
    const float* nt_emb   = (const float*)d_in[3];
    const float* root_emb = (const float*)d_in[4];
    const float* rule_w   = (const float*)d_in[5];
    const float* rule_b   = (const float*)d_in[6];
    const float* r_w_in = (const float*)d_in[7];  const float* r_b_in = (const float*)d_in[8];
    const float* r_w11  = (const float*)d_in[9];  const float* r_b11  = (const float*)d_in[10];
    const float* r_w12  = (const float*)d_in[11]; const float* r_b12  = (const float*)d_in[12];
    const float* r_w21  = (const float*)d_in[13]; const float* r_b21  = (const float*)d_in[14];
    const float* r_w22  = (const float*)d_in[15]; const float* r_b22  = (const float*)d_in[16];
    const float* r_wout = (const float*)d_in[17]; const float* r_bout = (const float*)d_in[18];
    const float* v_w_in = (const float*)d_in[19]; const float* v_b_in = (const float*)d_in[20];
    const float* v_w11  = (const float*)d_in[21]; const float* v_b11  = (const float*)d_in[22];
    const float* v_w12  = (const float*)d_in[23]; const float* v_b12  = (const float*)d_in[24];
    const float* v_w21  = (const float*)d_in[25]; const float* v_b21  = (const float*)d_in[26];
    const float* v_w22  = (const float*)d_in[27]; const float* v_b22  = (const float*)d_in[28];
    const float* v_wout = (const float*)d_in[29]; const float* v_bout = (const float*)d_in[30];
    float* out = (float*)d_out;

    char* ws = (char*)d_ws;
    size_t off = 0;
    auto alloc = [&](size_t bytes) -> char* {
        char* p = ws + off;
        off += (bytes + 255) & ~(size_t)255;
        return p;
    };
    // bf16 weights
    unsigned short* wb_vin  = (unsigned short*)alloc((size_t)DD * DZK * 2);
    unsigned short* wb_v11  = (unsigned short*)alloc((size_t)DD * DD * 2);
    unsigned short* wb_v12  = (unsigned short*)alloc((size_t)DD * DD * 2);
    unsigned short* wb_v21  = (unsigned short*)alloc((size_t)DD * DD * 2);
    unsigned short* wb_v22  = (unsigned short*)alloc((size_t)DD * DD * 2);
    unsigned short* wb_vout = (unsigned short*)alloc((size_t)VVP * DD * 2);
    unsigned short* wb_rin  = (unsigned short*)alloc((size_t)DD * DZK * 2);
    unsigned short* wb_r11  = (unsigned short*)alloc((size_t)DD * DD * 2);
    unsigned short* wb_r12  = (unsigned short*)alloc((size_t)DD * DD * 2);
    unsigned short* wb_r21  = (unsigned short*)alloc((size_t)DD * DD * 2);
    unsigned short* wb_r22  = (unsigned short*)alloc((size_t)DD * DD * 2);
    unsigned short* wb_rout = (unsigned short*)alloc((size_t)32 * DD * 2);
    unsigned short* wb_rule = (unsigned short*)alloc((size_t)S2P * DZK * 2);
    // inputs
    unsigned short* Xv = (unsigned short*)alloc((size_t)BB * TTC * DZK * 2);
    unsigned short* Xn = (unsigned short*)alloc((size_t)BB * NTC * DZK * 2);
    unsigned short* Xr = (unsigned short*)alloc((size_t)BB * DZK * 2);
    // activations (vocab M=960, root M=16)
    float* hA = (float*)alloc((size_t)960 * DD * 4);
    float* hB = (float*)alloc((size_t)960 * DD * 4);
    float* hT = (float*)alloc((size_t)960 * DD * 4);
    unsigned short* hAb = (unsigned short*)alloc((size_t)960 * DD * 2);
    unsigned short* hBb = (unsigned short*)alloc((size_t)960 * DD * 2);
    unsigned short* hTb = (unsigned short*)alloc((size_t)960 * DD * 2);
    float* rA = (float*)alloc((size_t)BB * DD * 4);
    float* rB = (float*)alloc((size_t)BB * DD * 4);
    float* rT = (float*)alloc((size_t)BB * DD * 4);
    unsigned short* rAb = (unsigned short*)alloc((size_t)BB * DD * 2);
    unsigned short* rBb = (unsigned short*)alloc((size_t)BB * DD * 2);
    unsigned short* rTb = (unsigned short*)alloc((size_t)BB * DD * 2);
    // logits / stats
    float* logv    = (float*)alloc((size_t)960 * VVP * 4);
    float* logr    = (float*)alloc((size_t)BB * 32 * 4);
    float* logrule = (float*)alloc((size_t)BB * NTC * S2P * 4);
    float* maxv = (float*)alloc(960 * 4);   float* lsev = (float*)alloc(960 * 4);
    float* maxr = (float*)alloc(BB * 4);    float* lser = (float*)alloc(BB * 4);
    float* maxru = (float*)alloc(BB * NTC * 4); float* lseru = (float*)alloc(BB * NTC * 4);
    float* km    = (float*)alloc(BB * NTC * 4);
    float* biasRule = (float*)alloc(S2P * 4);
    float* biasRout = (float*)alloc(32 * 4);
    float* biasVout = (float*)alloc(VVP * 4);
    float* unary  = (float*)alloc((size_t)BB * NSEQ * TTC * 4);
    float* rootls = (float*)alloc((size_t)BB * 32 * 4);
    float* beta   = (float*)alloc((size_t)BB * NSEQ * NSEQ * SPAD * 4);
    // inside-loop scratch
    unsigned short* erule = (unsigned short*)alloc((size_t)BB * NTSP * SPAD * 2);
    unsigned short* EL = (unsigned short*)alloc((size_t)BB * RMAX * SPAD * 2);
    unsigned short* ER = (unsigned short*)alloc((size_t)BB * RMAX * SPAD * 2);
    float* lm   = (float*)alloc((size_t)BB * RMAX * 4);
    float* rm   = (float*)alloc((size_t)BB * RMAX * 4);
    float* out1 = (float*)alloc((size_t)BB * RMAX * NTSP * 4);

    auto pack = [&](const float* s, unsigned short* d, int Rs, int Cs, int Rp, int Cp) {
        long n = (long)Rp * Cp;
        k_pack<<<dim3((unsigned)((n + 255) / 256)), dim3(256), 0, stream>>>(s, d, Rs, Cs, Rp, Cp);
    };
    auto gemm = [&](const unsigned short* A, const unsigned short* Wt, const float* bias,
                    const float* res, float* C, unsigned short* Cb,
                    int M, int N, int K, int relu,
                    int batch, long sA, long sW, long sC) {
        if ((M & 31) == 0 && (N & 31) == 0) {
            int tiles = (M >> 5) * (N >> 5);
            dim3 grid((unsigned)((tiles + 7) / 8), (unsigned)batch);
            k_gemm32<<<grid, dim3(256), 0, stream>>>(A, Wt, bias, res, C, Cb, M, N, K, relu, sA, sW, sC);
        } else {
            int tiles = (M >> 4) * (N >> 4);
            dim3 grid((unsigned)((tiles + 7) / 8), (unsigned)batch);
            k_gemm16<<<grid, dim3(256), 0, stream>>>(A, Wt, bias, res, C, Cb, M, N, K, relu, sA, sW, sC);
        }
    };

    // ---- pack weights / biases / inputs ----
    pack(v_w_in, wb_vin, DD, DZK, DD, DZK);
    pack(v_w11, wb_v11, DD, DD, DD, DD);   pack(v_w12, wb_v12, DD, DD, DD, DD);
    pack(v_w21, wb_v21, DD, DD, DD, DD);   pack(v_w22, wb_v22, DD, DD, DD, DD);
    pack(v_wout, wb_vout, VV, DD, VVP, DD);
    pack(r_w_in, wb_rin, DD, DZK, DD, DZK);
    pack(r_w11, wb_r11, DD, DD, DD, DD);   pack(r_w12, wb_r12, DD, DD, DD, DD);
    pack(r_w21, wb_r21, DD, DD, DD, DD);   pack(r_w22, wb_r22, DD, DD, DD, DD);
    pack(r_wout, wb_rout, NTC, DD, 32, DD);
    pack(rule_w, wb_rule, S2, DZK, S2P, DZK);
    k_padf<<<dim3((S2P + 255) / 256), dim3(256), 0, stream>>>(rule_b, biasRule, S2, S2P);
    k_padf<<<dim3(1), dim3(256), 0, stream>>>(r_bout, biasRout, NTC, 32);
    k_padf<<<dim3((VVP + 255) / 256), dim3(256), 0, stream>>>(v_bout, biasVout, VV, VVP);
    k_build_x<<<dim3((unsigned)(((long)BB * TTC * DZK + 255) / 256)), dim3(256), 0, stream>>>(t_emb, z, Xv, TTC);
    k_build_x<<<dim3((unsigned)(((long)BB * NTC * DZK + 255) / 256)), dim3(256), 0, stream>>>(nt_emb, z, Xn, NTC);
    k_build_x<<<dim3((unsigned)(((long)BB * DZK + 255) / 256)), dim3(256), 0, stream>>>(root_emb, z, Xr, 1);

    // ---- root MLP (M=16, 16x16 path) ----
    gemm(Xr,  wb_rin,  r_b_in, nullptr, rA, rAb, BB, DD, DZK, 0, 1, 0, 0, 0);
    gemm(rAb, wb_r11,  r_b11,  nullptr, rT, rTb, BB, DD, DD, 1, 1, 0, 0, 0);
    gemm(rTb, wb_r12,  r_b12,  rA,      rB, rBb, BB, DD, DD, 1, 1, 0, 0, 0);
    gemm(rBb, wb_r21,  r_b21,  nullptr, rT, rTb, BB, DD, DD, 1, 1, 0, 0, 0);
    gemm(rTb, wb_r22,  r_b22,  rB,      rA, rAb, BB, DD, DD, 1, 1, 0, 0, 0);
    gemm(rAb, wb_rout, biasRout, nullptr, logr, nullptr, BB, 32, DD, 0, 1, 0, 0, 0);

    // ---- vocab MLP (M=960, 32x32 path) ----
    gemm(Xv,  wb_vin,  v_b_in, nullptr, hA, hAb, 960, DD, DZK, 0, 1, 0, 0, 0);
    gemm(hAb, wb_v11,  v_b11,  nullptr, hT, hTb, 960, DD, DD, 1, 1, 0, 0, 0);
    gemm(hTb, wb_v12,  v_b12,  hA,      hB, hBb, 960, DD, DD, 1, 1, 0, 0, 0);
    gemm(hBb, wb_v21,  v_b21,  nullptr, hT, hTb, 960, DD, DD, 1, 1, 0, 0, 0);
    gemm(hTb, wb_v22,  v_b22,  hB,      hA, hAb, 960, DD, DD, 1, 1, 0, 0, 0);
    gemm(hAb, wb_vout, biasVout, nullptr, logv, nullptr, 960, VVP, DD, 0, 1, 0, 0, 0);

    // ---- rule head (M=480, N=8128, K=320, 32x32 path) ----
    gemm(Xn, wb_rule, biasRule, nullptr, logrule, nullptr, BB * NTC, S2P, DZK, 0, 1, 0, 0, 0);

    // ---- softmax stats / gathers ----
    k_rowstat<<<dim3(960), dim3(256), 0, stream>>>(logv, maxv, lsev, VV, VVP);
    k_rowstat<<<dim3(BB), dim3(256), 0, stream>>>(logr, maxr, lser, NTC, 32);
    k_rowstat<<<dim3(BB * NTC), dim3(256), 0, stream>>>(logrule, maxru, lseru, S2, S2P);
    k_unary<<<dim3((BB * NSEQ * TTC + 255) / 256), dim3(256), 0, stream>>>(logv, lsev, trg, unary);
    k_rootls<<<dim3((BB * 32 + 255) / 256), dim3(256), 0, stream>>>(logr, lser, rootls);
    k_km<<<dim3((BB * NTC + 255) / 256), dim3(256), 0, stream>>>(maxru, lseru, km);

    // ---- erule pack (zero pads then fill) ----
    {
        long n = (long)BB * NTSP * SPAD;
        k_zero16<<<dim3((unsigned)((n + 255) / 256)), dim3(256), 0, stream>>>(erule, n);
        long t = (long)BB * NTC * S2;
        k_erule<<<dim3((unsigned)((t + 255) / 256)), dim3(256), 0, stream>>>(logrule, maxru, erule);
    }

    // ---- beta init ----
    {
        long n = (long)BB * NSEQ * NSEQ * SPAD;
        k_beta_fill<<<dim3((unsigned)((n + 255) / 256)), dim3(256), 0, stream>>>(beta);
        k_beta_diag<<<dim3((BB * NSEQ * TTC + 255) / 256), dim3(256), 0, stream>>>(unary, beta);
    }

    // ---- inside algorithm, widths 2..n ----
    for (int w = 2; w <= NSEQ; ++w) {
        int ns = NSEQ - w + 1;
        int rowsReal = ns * (w - 1);
        int rowsPad = (rowsReal + 31) & ~31;
        k_prep<<<dim3((BB * rowsPad + 255) / 256), dim3(256), 0, stream>>>(beta, EL, ER, lm, rm, w, rowsPad);
        // stage-1: Out1[b][row][k*S+l] = sum_r ER[b][row][r] * erule[b][(k,l)][r]
        gemm(ER, erule, nullptr, nullptr, out1, nullptr,
             rowsPad, NTSP, SPAD, 0,
             BB, (long)rowsPad * SPAD, (long)NTSP * SPAD, (long)rowsPad * NTSP);
        // stage-2: contract with EL, log, logsumexp over splits, write beta
        k_combine<<<dim3(BB * ns), dim3(256), 0, stream>>>(EL, lm, rm, km, out1, beta, w, rowsPad);
    }

    // ---- final: out = -logsumexp(beta[:,0,n-1,:NT] + root), kl = 0 ----
    k_final<<<dim3(1), dim3(32), 0, stream>>>(beta, rootls, out);
}
